// EncoderLayer_71073118814662
// MI455X (gfx1250) — matrix operand order repaired and checked by reading, not compiled, not run on hardware
//
#include <hip/hip_runtime.h>
#include <hip/hip_bf16.h>
#include <math.h>

// ---------------------------------------------------------------------------
// EncoderLayer for MI455X (gfx1250): bf16 WMMA GEMMs (async-LDS double
// buffered, 32x64 wave tiles) + fused-bias attention.
// ---------------------------------------------------------------------------

typedef __bf16 bf16_t;
typedef __attribute__((ext_vector_type(16))) __bf16 v16bf;
typedef __attribute__((ext_vector_type(8)))  float  v8f;

#define NHEADS 32
#define DKH    32
#define SEQ    513
#define NBATCH 8
#define HID    1024
#define FFN    4096
#define MROWS  (NBATCH * SEQ)   // 4104

// Build a v16bf A/B fragment from two 16-byte chunks (contiguous k order,
// k = 16*half + s: elements 0..7 = low K-octet, 8..15 = high K-octet).
__device__ __forceinline__ v16bf frag_from(uint4 lo, uint4 hi) {
    union { unsigned u[8]; v16bf v; } t;
    t.u[0] = lo.x; t.u[1] = lo.y; t.u[2] = lo.z; t.u[3] = lo.w;
    t.u[4] = hi.x; t.u[5] = hi.y; t.u[6] = hi.z; t.u[7] = hi.w;
    return t.v;
}

__device__ __forceinline__ v8f wmma_bf16(v16bf a, v16bf b, v8f c) {
    return __builtin_amdgcn_wmma_f32_16x16x32_bf16(
        /*neg_a=*/false, a, /*neg_b=*/false, b,
        /*c_mod=*/(short)0, c, /*reuse_a=*/false, /*reuse_b=*/false);
}

__device__ __forceinline__ v8f v8f_zero() {
    v8f z = {0.f, 0.f, 0.f, 0.f, 0.f, 0.f, 0.f, 0.f};
    return z;
}

// Async global->LDS DMA (tracked by ASYNCcnt). LDS destination address is the
// low 32 bits of the generic pointer (ISA 10.2: flat addr truncates to DS
// offset inside the LDS aperture).
__device__ __forceinline__ void async_ld_b128(void* lds_dst, const void* gsrc) {
    unsigned dst = (unsigned)(unsigned long long)lds_dst;
    asm volatile("global_load_async_to_lds_b128 %0, %1, off"
                 :: "v"(dst), "v"(gsrc) : "memory");
}
__device__ __forceinline__ void wait_async0() {
    asm volatile("s_wait_asynccnt 0" ::: "memory");
}

// ---------------------------------------------------------------------------
// LayerNorm: one block (128 threads) per row of 1024; writes bf16.
// ---------------------------------------------------------------------------
__global__ __launch_bounds__(128)
void ln_kernel(const float* __restrict__ x, const float* __restrict__ g,
               const float* __restrict__ bta, bf16_t* __restrict__ y) {
    int row = blockIdx.x;
    const float* xr = x + (size_t)row * HID;
    float v[8];
    float s = 0.f, s2 = 0.f;
#pragma unroll
    for (int k = 0; k < 8; ++k) {
        float t = xr[threadIdx.x + 128 * k];
        v[k] = t; s += t; s2 += t * t;
    }
#pragma unroll
    for (int o = 16; o >= 1; o >>= 1) {
        s  += __shfl_xor(s,  o, 32);
        s2 += __shfl_xor(s2, o, 32);
    }
    __shared__ float sh[8];
    int wave = threadIdx.x >> 5, lane = threadIdx.x & 31;
    if (lane == 0) { sh[wave] = s; sh[4 + wave] = s2; }
    __syncthreads();
    s  = sh[0] + sh[1] + sh[2] + sh[3];
    s2 = sh[4] + sh[5] + sh[6] + sh[7];
    float mean = s * (1.0f / HID);
    float var  = s2 * (1.0f / HID) - mean * mean;
    float rstd = rsqrtf(var + 1e-5f);
#pragma unroll
    for (int k = 0; k < 8; ++k) {
        int c = threadIdx.x + 128 * k;
        float o = (v[k] - mean) * rstd * g[c] + bta[c];
        y[(size_t)row * HID + c] = (bf16_t)o;
    }
}

// ---------------------------------------------------------------------------
// Generic bf16 WMMA GEMM: C[M,N] = A[M,K](bf16) * B[K,N](f32->bf16) + epilogue
//   mode 0: qkv projection -> ((b*H + h)*S + s)*32 + d layout, * scale, bf16
//   mode 2: out-proj: f32 out = acc + bias + resid
//   mode 3: FFN1: bf16 out = gelu(acc + bias)
//   mode 4: FFN2: f32 out = acc + bias + resid
// Block tile 64(M)x128(N), 4 waves in 2x2, each wave a 32x64 tile (2x4 WMMA).
// A tile: async global->LDS DMA, double buffered (ASYNCcnt).
// B tile: f32 global loads (2k x 4n sub-blocks) -> packed bf16x2 LDS stores.
// N must be a multiple of 128 (holds: 1024 and 4096).
// ---------------------------------------------------------------------------
#define GTM 64
#define GTN 128
#define GTK 32
#define LDAP 40   // padded LDS row stride (elements); 80B keeps 16B alignment

__global__ __launch_bounds__(128)
void gemm_kernel(const bf16_t* __restrict__ A, const float* __restrict__ Bm,
                 const float* __restrict__ bias, const float* __restrict__ resid,
                 void* __restrict__ outp, int M, int N, int K, int mode,
                 float scale) {
    __shared__ bf16_t As[2][GTM * LDAP];
    __shared__ bf16_t Bt[2][GTN * LDAP];
    int tid  = threadIdx.x;
    int wave = tid >> 5, lane = tid & 31;
    int wm = wave >> 1, wn = wave & 1;
    int m0 = blockIdx.y * GTM, n0 = blockIdx.x * GTN;
    int mrow = lane & 15, half = lane >> 4;

    v8f acc[2][4];
#pragma unroll
    for (int i = 0; i < 2; ++i)
#pragma unroll
        for (int j = 0; j < 4; ++j) acc[i][j] = v8f_zero();

    // --- staging helpers -------------------------------------------------
    // A tile (64x32 bf16): 256 x 16B chunks, 2 per thread, async DMA to LDS.
    // Out-of-range rows are clamped to M-1 (their C rows are never stored),
    // which keeps the copy loop branch-free.
    auto issue_A = [&](int k0, int buf) {
#pragma unroll
        for (int t = 0; t < 2; ++t) {
            int c = tid + t * 128;
            int r = c >> 2, cc = (c & 3) * 8;
            int gm = m0 + r;
            if (gm >= M) gm = M - 1;
            async_ld_b128(As[buf] + r * LDAP + cc,
                          A + (size_t)gm * K + k0 + cc);
        }
    };
    // B tile (32k x 128n f32): each thread owns four 2k x 4n sub-blocks.
    // sub-block c: kp = c>>5 (k = 2*kp), ng = c&31 (n = 4*ng).
    auto load_B = [&](int k0, float4* regs) {
#pragma unroll
        for (int t = 0; t < 4; ++t) {
            int c = tid + t * 128;
            int kk = (c >> 5) * 2, nn = (c & 31) * 4;
            const float* src = Bm + (size_t)(k0 + kk) * N + n0 + nn;
            regs[2 * t]     = *(const float4*)src;
            regs[2 * t + 1] = *(const float4*)(src + N);
        }
    };
    auto store_B = [&](const float4* regs, int buf) {
#pragma unroll
        for (int t = 0; t < 4; ++t) {
            int c = tid + t * 128;
            int kk = (c >> 5) * 2, nn = (c & 31) * 4;
            bf16_t* dst = Bt[buf];
            const float* r0 = &regs[2 * t].x;
            const float* r1 = &regs[2 * t + 1].x;
#pragma unroll
            for (int e = 0; e < 4; ++e) {
                union { bf16_t h[2]; unsigned u; } pk;
                pk.h[0] = (bf16_t)r0[e];   // k
                pk.h[1] = (bf16_t)r1[e];   // k+1
                *(unsigned*)(dst + (nn + e) * LDAP + kk) = pk.u;
            }
        }
    };

    // --- prologue: fill buffer 0 ----------------------------------------
    float4 breg[8];
    issue_A(0, 0);
    load_B(0, breg);
    store_B(breg, 0);
    wait_async0();
    __syncthreads();

    int p = 0;
    for (int k0 = 0; k0 < K; k0 += GTK) {
        bool has_next = (k0 + GTK) < K;
        if (has_next) {
            issue_A(k0 + GTK, p ^ 1);   // DMA next A tile while we compute
            load_B(k0 + GTK, breg);     // next B tile in flight
        }

        v16bf af[2], bfr[4];
#pragma unroll
        for (int i = 0; i < 2; ++i) {
            const bf16_t* q = As[p] + (wm * 32 + i * 16 + mrow) * LDAP;
            af[i] = frag_from(*(const uint4*)(q + 16 * half),
                              *(const uint4*)(q + 16 * half + 8));
        }
#pragma unroll
        for (int j = 0; j < 4; ++j) {
            const bf16_t* q = Bt[p] + (wn * 64 + j * 16 + mrow) * LDAP + 16 * half;
            bfr[j] = frag_from(*(const uint4*)q, *(const uint4*)(q + 8));
        }
#pragma unroll
        for (int i = 0; i < 2; ++i)
#pragma unroll
            for (int j = 0; j < 4; ++j)
                acc[i][j] = wmma_bf16(af[i], bfr[j], acc[i][j]);

        if (has_next) store_B(breg, p ^ 1);
        wait_async0();
        __syncthreads();
        p ^= 1;
    }

    // --- epilogue ---------------------------------------------------------
#pragma unroll
    for (int i = 0; i < 2; ++i)
#pragma unroll
        for (int j = 0; j < 4; ++j)
#pragma unroll
            for (int r = 0; r < 8; ++r) {
                int gm = m0 + wm * 32 + i * 16 + r + 8 * half;
                int gn = n0 + wn * 64 + j * 16 + mrow;
                if (gm >= M) continue;
                float v = acc[i][j][r] + bias[gn];
                if (mode == 0) {
                    v *= scale;
                    int bb = gm / SEQ, ss = gm % SEQ;
                    int hh = gn >> 5, dd = gn & 31;
                    ((bf16_t*)outp)[(((size_t)(bb * NHEADS + hh) * SEQ + ss) << 5) + dd] =
                        (bf16_t)v;
                } else if (mode == 2) {
                    ((float*)outp)[(size_t)gm * N + gn] =
                        v + resid[(size_t)gm * N + gn];
                } else if (mode == 3) {
                    float ge = 0.5f * v * (1.0f + erff(v * 0.70710678118f));
                    ((bf16_t*)outp)[(size_t)gm * N + gn] = (bf16_t)ge;
                } else {
                    ((float*)outp)[(size_t)gm * N + gn] =
                        v + resid[(size_t)gm * N + gn];
                }
            }
}

// ---------------------------------------------------------------------------
// Attention: one wave per (q-tile of 16 rows, head, batch).
//   scores = q*k^T via WMMA (A = q 16x32 bf16, B = k^T direct from global),
//   rel-pos bias fixup fused (single pass over attn_bias), softmax in LDS,
//   P@V via WMMA with V chunks transposed through LDS.
// ---------------------------------------------------------------------------
#define NKT   33                 // ceil(513/16)
#define SCOLS (NKT * 16)         // 528
#define SLD   544                // padded f32 row stride

__global__ __launch_bounds__(32)
void attn_kernel(const bf16_t* __restrict__ qg, const bf16_t* __restrict__ kg,
                 const bf16_t* __restrict__ vg,
                 const float* __restrict__ attn_bias,
                 const int*   __restrict__ rel_pos,
                 const float* __restrict__ rel_emb,
                 bf16_t* __restrict__ og) {
    __shared__ float  Sld[16 * SLD];
    __shared__ bf16_t Vt[32 * 40];

    int qt = blockIdx.x, h = blockIdx.y, b = blockIdx.z;
    int lane = threadIdx.x;
    int mrow = lane & 15, half = lane >> 4;
    int i0 = qt * 16;
    size_t headbase = (size_t)(b * NHEADS + h) * SEQ;

    // q A-fragment (16 rows x 32 dk), loaded once
    v16bf qa;
    {
        uint4 lo = make_uint4(0u, 0u, 0u, 0u), hi = lo;
        int gi = i0 + mrow;
        if (gi < SEQ) {
            const bf16_t* p = qg + ((headbase + gi) << 5);
            lo = *(const uint4*)(p + 16 * half);
            hi = *(const uint4*)(p + 16 * half + 8);
        }
        qa = frag_from(lo, hi);
    }

    const float* biasbase = attn_bias + (size_t)(b * NHEADS + h) * SEQ * SEQ;

    // ---- scores + fused bias fixup -> LDS ----
    for (int kt = 0; kt < NKT; ++kt) {
        int j0 = kt * 16;
        v16bf kb;
        {
            uint4 lo = make_uint4(0u, 0u, 0u, 0u), hi = lo;
            int gj = j0 + mrow;
            if (gj < SEQ) {
                const bf16_t* p = kg + ((headbase + gj) << 5) + 16 * half;
                lo = *(const uint4*)p;
                hi = *(const uint4*)(p + 8);
            }
            kb = frag_from(lo, hi);
        }
        v8f c = wmma_bf16(qa, kb, v8f_zero());

        int jj = j0 + mrow;
#pragma unroll
        for (int r = 0; r < 8; ++r) {
            int ii = i0 + r + 8 * half;
            float val;
            if (ii < SEQ && jj < SEQ) {
                float ab = biasbase[(size_t)ii * SEQ + jj];
                if (ii >= 1 && jj >= 1) {
                    int rp = rel_pos[((size_t)b * (SEQ - 1) + (ii - 1)) * (SEQ - 1)
                                     + (jj - 1)];
                    float inner = ab + rel_emb[rp * NHEADS + h];
                    bool keep = ((float)rp <= 15.5f) || (inner < -1e8f);
                    val = keep ? inner : 0.0f;
                } else {
                    val = ab;
                }
                val += c[r];
            } else {
                val = -1e30f;   // pad columns -> exp() == 0
            }
            Sld[(r + 8 * half) * SLD + j0 + mrow] = val;
        }
    }
    __syncthreads();

    // ---- softmax (lane pair {L, L+16} shares row L%16) ----
    float mx = -3.0e38f;
    for (int c = half; c < SCOLS; c += 2)
        mx = fmaxf(mx, Sld[mrow * SLD + c]);
    mx = fmaxf(mx, __shfl_xor(mx, 16, 32));
    float sum = 0.f;
    for (int c = half; c < SCOLS; c += 2) {
        float p = __expf(Sld[mrow * SLD + c] - mx);
        Sld[mrow * SLD + c] = p;
        sum += p;
    }
    sum += __shfl_xor(sum, 16, 32);
    float inv = 1.0f / sum;   // normalization folded into P fragments
    __syncthreads();

    // ---- O = P @ V (17 K-chunks of 32) ----
    v8f acc0 = v8f_zero(), acc1 = v8f_zero();
    for (int ct = 0; ct < 17; ++ct) {
        int kkb = ct * 32;
        {   // stage V chunk transposed: Vt[d][kk]
            int gk = kkb + lane;
            if (gk < SEQ) {
                const bf16_t* p = vg + ((headbase + gk) << 5);
#pragma unroll
                for (int n = 0; n < 32; ++n) Vt[n * 40 + lane] = p[n];
            } else {
#pragma unroll
                for (int n = 0; n < 32; ++n) Vt[n * 40 + lane] = (bf16_t)0.0f;
            }
        }
        __syncthreads();

        // P A-fragment (row mrow, cols kkb + 16*half .. kkb + 16*half + 15)
        v16bf pf;
        {
            union { bf16_t e[16]; v16bf v; } t;
            const float* prow = Sld + mrow * SLD + kkb;
#pragma unroll
            for (int e = 0; e < 8; ++e)
                t.e[e] = (bf16_t)(prow[16 * half + e] * inv);
#pragma unroll
            for (int e = 0; e < 8; ++e)
                t.e[8 + e] = (bf16_t)(prow[16 * half + 8 + e] * inv);
            pf = t.v;
        }
        const bf16_t* p0 = Vt + (mrow)*40 + 16 * half;
        const bf16_t* p1 = Vt + (16 + mrow) * 40 + 16 * half;
        v16bf vf0 = frag_from(*(const uint4*)p0, *(const uint4*)(p0 + 8));
        v16bf vf1 = frag_from(*(const uint4*)p1, *(const uint4*)(p1 + 8));
        acc0 = wmma_bf16(pf, vf0, acc0);
        acc1 = wmma_bf16(pf, vf1, acc1);
        __syncthreads();
    }

    // ---- write O as bf16 [B,S,H] ----
#pragma unroll
    for (int r = 0; r < 8; ++r) {
        int gi = i0 + r + 8 * half;
        if (gi < SEQ) {
            size_t base = ((size_t)(b * SEQ + gi)) * HID + h * DKH;
            og[base + mrow]      = (bf16_t)acc0[r];
            og[base + 16 + mrow] = (bf16_t)acc1[r];
        }
    }
}

// ---------------------------------------------------------------------------
// Host orchestration
// ---------------------------------------------------------------------------
extern "C" void kernel_launch(void* const* d_in, const int* in_sizes, int n_in,
                              void* d_out, int out_size, void* d_ws, size_t ws_size,
                              hipStream_t stream) {
    (void)in_sizes; (void)n_in; (void)out_size; (void)ws_size;
    const float* x         = (const float*)d_in[0];
    const float* attn_bias = (const float*)d_in[1];
    const float* rel_emb   = (const float*)d_in[2];
    const float* Wq = (const float*)d_in[3];  const float* bq = (const float*)d_in[4];
    const float* Wk = (const float*)d_in[5];  const float* bk = (const float*)d_in[6];
    const float* Wv = (const float*)d_in[7];  const float* bv = (const float*)d_in[8];
    const float* Wo = (const float*)d_in[9];  const float* bo = (const float*)d_in[10];
    const float* W1 = (const float*)d_in[11]; const float* b1 = (const float*)d_in[12];
    const float* W2 = (const float*)d_in[13]; const float* b2 = (const float*)d_in[14];
    const float* ln1_g = (const float*)d_in[15]; const float* ln1_b = (const float*)d_in[16];
    const float* ln2_g = (const float*)d_in[17]; const float* ln2_b = (const float*)d_in[18];
    const int*   rel_pos = (const int*)d_in[19];

    // workspace carve-up
    char* ws = (char*)d_ws;
    size_t off = 0;
    auto take = [&](size_t bytes) -> char* {
        char* p = ws + off;
        off += (bytes + 255) & ~(size_t)255;
        return p;
    };
    bf16_t* y1 = (bf16_t*)take((size_t)MROWS * HID * 2);
    bf16_t* qb = (bf16_t*)take((size_t)MROWS * HID * 2);
    bf16_t* kb = (bf16_t*)take((size_t)MROWS * HID * 2);
    bf16_t* vb = (bf16_t*)take((size_t)MROWS * HID * 2);
    bf16_t* ob = (bf16_t*)take((size_t)MROWS * HID * 2);
    float*  x1 = (float*)take((size_t)MROWS * HID * 4);
    bf16_t* y2 = (bf16_t*)take((size_t)MROWS * HID * 2);
    bf16_t* h1 = (bf16_t*)take((size_t)MROWS * FFN * 2);

    const int mtiles = (MROWS + GTM - 1) / GTM;   // 65
    const float qscale = 0.17677669529663687f;    // 1/sqrt(32)

    // LN1
    ln_kernel<<<MROWS, 128, 0, stream>>>(x, ln1_g, ln1_b, y1);
    // QKV projections (bf16, qkv head layout)
    gemm_kernel<<<dim3(HID / GTN, mtiles), 128, 0, stream>>>(
        y1, Wq, bq, nullptr, qb, MROWS, HID, HID, 0, qscale);
    gemm_kernel<<<dim3(HID / GTN, mtiles), 128, 0, stream>>>(
        y1, Wk, bk, nullptr, kb, MROWS, HID, HID, 0, 1.0f);
    gemm_kernel<<<dim3(HID / GTN, mtiles), 128, 0, stream>>>(
        y1, Wv, bv, nullptr, vb, MROWS, HID, HID, 0, 1.0f);
    // attention with fused rel-pos bias fixup
    attn_kernel<<<dim3(NKT, NHEADS, NBATCH), 32, 0, stream>>>(
        qb, kb, vb, attn_bias, rel_pos, rel_emb, ob);
    // output projection + residual -> x1 (f32)
    gemm_kernel<<<dim3(HID / GTN, mtiles), 128, 0, stream>>>(
        ob, Wo, bo, x, x1, MROWS, HID, HID, 2, 1.0f);
    // LN2
    ln_kernel<<<MROWS, 128, 0, stream>>>(x1, ln2_g, ln2_b, y2);
    // FFN1 + exact GELU (bf16)
    gemm_kernel<<<dim3(FFN / GTN, mtiles), 128, 0, stream>>>(
        y2, W1, b1, nullptr, h1, MROWS, FFN, HID, 3, 1.0f);
    // FFN2 + residual -> d_out (f32)
    gemm_kernel<<<dim3(HID / GTN, mtiles), 128, 0, stream>>>(
        h1, W2, b2, x1, d_out, MROWS, HID, FFN, 4, 1.0f);
}
